// StandardAttention_25769804348
// MI455X (gfx1250) — matrix operand order repaired and checked by reading, not compiled, not run on hardware
//
#include <hip/hip_runtime.h>
#include <hip/hip_bf16.h>

// ---------------------------------------------------------------------------
// Problem constants (from reference)
// ---------------------------------------------------------------------------
#define B_    4
#define T_    1024
#define DM_   4096
#define H_    32
#define KVH_  8
#define HD_   128
#define NTOK_ (B_ * T_)              // 4096
#define NBLK_PER_SEQ_ (T_ / 16)      // 64
#define QK_SCALE 0.08838834764831845f  // 1/sqrt(128)

typedef __attribute__((ext_vector_type(2)))  float  v2f;
typedef __attribute__((ext_vector_type(8)))  float  v8f;
typedef __attribute__((ext_vector_type(4)))  unsigned v4u;
typedef __attribute__((ext_vector_type(16))) __bf16 v16bf;

union FragU {
  v4u   q[2];
  v16bf v;
};

__device__ __forceinline__ v8f wmma_f32(v2f a, v2f b, v8f c) {
  return __builtin_amdgcn_wmma_f32_16x16x4_f32(false, a, false, b, (short)0, c,
                                               false, false);
}
__device__ __forceinline__ v8f wmma_bf16(v16bf a, v16bf b, v8f c) {
  return __builtin_amdgcn_wmma_f32_16x16x32_bf16(false, a, false, b, (short)0,
                                                 c, false, false);
}
__device__ __forceinline__ v8f v8zero() {
  v8f z = {0.f, 0.f, 0.f, 0.f, 0.f, 0.f, 0.f, 0.f};
  return z;
}

// f32 -> bf16 (RNE), returning the bf16 bits and the f32 residual
__device__ __forceinline__ unsigned bf16_rne(float x, float* resid) {
  unsigned u  = __float_as_uint(x);
  unsigned hv = (u + 0x7FFFu + ((u >> 16) & 1u)) >> 16;
  *resid = x - __uint_as_float(hv << 16);
  return hv;
}
__device__ __forceinline__ unsigned bf16_rne_only(float x) {
  unsigned u = __float_as_uint(x);
  return (u + 0x7FFFu + ((u >> 16) & 1u)) >> 16;
}

// async 16B global -> LDS copy (per-lane), tracked by ASYNCcnt
__device__ __forceinline__ void async_b128(unsigned lds_off, const void* g) {
  asm volatile("global_load_async_to_lds_b128 %0, %1, off"
               :: "v"(lds_off), "v"(g) : "memory");
}

// ---------------------------------------------------------------------------
// GEMM  C[M,N] = A[M,K] @ B[K,N]  (f32 in/out), computed as bf16 hi/lo split:
//   A*B ~= Ah*Bh + Ah*Bl + Al*Bh   (f32 accumulate, ~1e-5 rel error)
// Block = 256 thr (8 waves), tile 128x128, K-chunk 32.
// Wave w owns rows [w*16, w*16+16) x all 128 cols -> 8 C tiles, 24 bf16 WMMA
// per chunk. LDS tiles stored in fragment-native kpair-dword layout, padded
// to 20 dwords/row (conflict-free b128 fragment loads).
// ---------------------------------------------------------------------------
#define GM 128
#define GN 128
#define GK 32
#define LPAD 20  // dwords per LDS row (16 kpairs + 4 pad)

__global__ __launch_bounds__(256) void gemm_bf16x3(
    const float* __restrict__ A, const float* __restrict__ B,
    float* __restrict__ C, int M, int N, int K) {
  __shared__ unsigned AsHi[GM][LPAD];
  __shared__ unsigned AsLo[GM][LPAD];
  __shared__ unsigned BsHi[GN][LPAD];
  __shared__ unsigned BsLo[GN][LPAD];

  const int tid  = threadIdx.x;
  const int wave = tid >> 5;
  const int lane = tid & 31;
  const int lo   = lane & 15;
  const int hi   = lane >> 4;

  const int row0 = blockIdx.x * GM;
  const int col0 = blockIdx.y * GN;

  // staging coordinates
  const int am = tid >> 1;            // A row 0..127
  const int ac = (tid & 1) * 8;       // A kpair dword base 0 or 8
  const int bkp = tid >> 4;           // B kpair 0..15
  const int bn0 = (tid & 15) * 8;     // B col base

  v8f acc[8];
#pragma unroll
  for (int i = 0; i < 8; ++i) acc[i] = v8zero();

  float fa[16];  // 16 consecutive k of one A row
  float fb[16];  // cols n0..n0+7 of rows 2kp and 2kp+1

  auto load_next = [&](int k0) {
    const float* ap = &A[(size_t)(row0 + am) * K + k0 + ac * 2];
    *reinterpret_cast<float4*>(&fa[0])  = *reinterpret_cast<const float4*>(ap);
    *reinterpret_cast<float4*>(&fa[4])  = *reinterpret_cast<const float4*>(ap + 4);
    *reinterpret_cast<float4*>(&fa[8])  = *reinterpret_cast<const float4*>(ap + 8);
    *reinterpret_cast<float4*>(&fa[12]) = *reinterpret_cast<const float4*>(ap + 12);
    const float* bp = &B[(size_t)(k0 + 2 * bkp) * N + col0 + bn0];
    *reinterpret_cast<float4*>(&fb[0])  = *reinterpret_cast<const float4*>(bp);
    *reinterpret_cast<float4*>(&fb[4])  = *reinterpret_cast<const float4*>(bp + 4);
    *reinterpret_cast<float4*>(&fb[8])  = *reinterpret_cast<const float4*>(bp + N);
    *reinterpret_cast<float4*>(&fb[12]) = *reinterpret_cast<const float4*>(bp + N + 4);
  };

  auto store_lds = [&]() {
    // A: 8 kpairs -> 2 uint4 hi + 2 uint4 lo
    unsigned h[8], l[8];
#pragma unroll
    for (int j = 0; j < 8; ++j) {
      float re, ro;
      unsigned he = bf16_rne(fa[2 * j], &re);
      unsigned ho = bf16_rne(fa[2 * j + 1], &ro);
      h[j] = he | (ho << 16);
      l[j] = bf16_rne_only(re) | (bf16_rne_only(ro) << 16);
    }
    *reinterpret_cast<v4u*>(&AsHi[am][ac])     = v4u{h[0], h[1], h[2], h[3]};
    *reinterpret_cast<v4u*>(&AsHi[am][ac + 4]) = v4u{h[4], h[5], h[6], h[7]};
    *reinterpret_cast<v4u*>(&AsLo[am][ac])     = v4u{l[0], l[1], l[2], l[3]};
    *reinterpret_cast<v4u*>(&AsLo[am][ac + 4]) = v4u{l[4], l[5], l[6], l[7]};
    // B: transpose into [col][kpair]
#pragma unroll
    for (int i = 0; i < 8; ++i) {
      float re, ro;
      unsigned he = bf16_rne(fb[i], &re);        // row 2kp
      unsigned ho = bf16_rne(fb[8 + i], &ro);    // row 2kp+1
      BsHi[bn0 + i][bkp] = he | (ho << 16);
      BsLo[bn0 + i][bkp] = bf16_rne_only(re) | (bf16_rne_only(ro) << 16);
    }
  };

  load_next(0);
  for (int k0 = 0; k0 < K; k0 += GK) {
    store_lds();
    __syncthreads();
    if (k0 + GK < K) load_next(k0 + GK);  // overlap with compute below

    const int m   = wave * 16 + lo;
    const int ha4 = hi * 8;
    FragU aH, aL;
    aH.q[0] = *reinterpret_cast<const v4u*>(&AsHi[m][ha4]);
    aH.q[1] = *reinterpret_cast<const v4u*>(&AsHi[m][ha4 + 4]);
    aL.q[0] = *reinterpret_cast<const v4u*>(&AsLo[m][ha4]);
    aL.q[1] = *reinterpret_cast<const v4u*>(&AsLo[m][ha4 + 4]);
#pragma unroll
    for (int nt = 0; nt < 8; ++nt) {
      const int n   = nt * 16 + lo;
      const int hb8 = hi * 8;
      FragU bH, bL;
      bH.q[0] = *reinterpret_cast<const v4u*>(&BsHi[n][hb8]);
      bH.q[1] = *reinterpret_cast<const v4u*>(&BsHi[n][hb8 + 4]);
      bL.q[0] = *reinterpret_cast<const v4u*>(&BsLo[n][hb8]);
      bL.q[1] = *reinterpret_cast<const v4u*>(&BsLo[n][hb8 + 4]);
      acc[nt] = wmma_bf16(aH.v, bH.v, acc[nt]);
      acc[nt] = wmma_bf16(aH.v, bL.v, acc[nt]);
      acc[nt] = wmma_bf16(aL.v, bH.v, acc[nt]);
    }
    __syncthreads();
  }

  const int hrow = hi << 3;
#pragma unroll
  for (int nt = 0; nt < 8; ++nt) {
    int col = col0 + nt * 16 + lo;
#pragma unroll
    for (int r = 0; r < 8; ++r) {
      int row = row0 + wave * 16 + r + hrow;
      C[(size_t)row * N + col] = acc[nt][r];
    }
  }
}

// ---------------------------------------------------------------------------
// RoPE on q in-place; RoPE k + scatter k/v into private paged cache.
// ---------------------------------------------------------------------------
__global__ __launch_bounds__(256) void rope_scatter(
    float* __restrict__ q, const float* __restrict__ kv,
    float* __restrict__ kws, float* __restrict__ vws,
    const float* __restrict__ cosb, const float* __restrict__ sinb,
    const int* __restrict__ positions, const int* __restrict__ btab) {
  const int row = blockIdx.x;
  const int tid = threadIdx.x;
  const int pos = positions[row];
  const float* cr = cosb + (size_t)pos * (HD_ / 2);
  const float* sr = sinb + (size_t)pos * (HD_ / 2);

  float* qr = q + (size_t)row * (H_ * HD_);
  for (int p = tid; p < H_ * HD_ / 2; p += 256) {
    int d = p & 63;
    float c = cr[d], s = sr[d];
    float x1 = qr[2 * p], x2 = qr[2 * p + 1];
    qr[2 * p]     = x1 * c - x2 * s;
    qr[2 * p + 1] = x1 * s + x2 * c;
  }

  const int b   = row / T_;
  const int blk = btab[b * NBLK_PER_SEQ_ + (pos >> 4)];
  const int off = pos & 15;
  const float* kr = kv + (size_t)row * (2 * KVH_ * HD_);
  const float* vr = kr + KVH_ * HD_;

  for (int p = tid; p < KVH_ * HD_ / 2; p += 256) {
    int kvh = p >> 6;
    int d   = p & 63;
    float c = cr[d], s = sr[d];
    float x1 = kr[2 * p], x2 = kr[2 * p + 1];
    float* dst = kws + ((((size_t)blk * KVH_ + kvh) * 16 + off) * HD_) + 2 * d;
    dst[0] = x1 * c - x2 * s;
    dst[1] = x1 * s + x2 * c;
  }
  for (int e = tid; e < KVH_ * HD_; e += 256) {
    int kvh = e >> 7;
    int d   = e & 127;
    vws[(((size_t)blk * KVH_ + kvh) * 16 + off) * HD_ + d] = vr[e];
  }
}

// ---------------------------------------------------------------------------
// Flash attention (exact f32 WMMA). Grid = B*H*(T/128); 8 waves, each owns a
// 16-row Q tile. K/V pages staged with double-buffered async-to-LDS copies.
// ---------------------------------------------------------------------------
#define KPAD (HD_ + 4)  // 132

__global__ __launch_bounds__(256) void attn_fa(
    const float* __restrict__ q, const float* __restrict__ kws,
    const float* __restrict__ vws, const int* __restrict__ btab,
    float* __restrict__ o) {
  __shared__ float Kt[2][16][KPAD];
  __shared__ float Vt[2][16][KPAD];
  __shared__ float Ps[8][16][20];

  const int tid  = threadIdx.x;
  const int wave = tid >> 5;
  const int lane = tid & 31;
  const int lo   = lane & 15;
  const int hi2  = (lane >> 4) << 1;
  const int hrow = (lane >> 4) << 3;

  const int qc  = blockIdx.x & 7;
  const int h   = (blockIdx.x >> 3) & 31;
  const int b   = blockIdx.x >> 8;
  const int kvh = h >> 2;
  const int qbase = qc * 128;

  // per-thread async staging coords: 2 x b128 per tensor per page
  const int cr0 = tid >> 5;           // rows 0..7   (it=0)
  const int cr1 = (tid >> 5) + 8;     // rows 8..15  (it=1)
  const int cc4 = (tid & 31) << 2;    // col dword 0..124
  const unsigned ktbase = (unsigned)(uintptr_t)&Kt[0][0][0];
  const unsigned vtbase = (unsigned)(uintptr_t)&Vt[0][0][0];

  auto issue_page = [&](int buf, int kb) {
    const int blk = btab[b * NBLK_PER_SEQ_ + kb];
    const float* ksrc = kws + (((size_t)blk * KVH_ + kvh) * 16) * HD_;
    const float* vsrc = vws + (((size_t)blk * KVH_ + kvh) * 16) * HD_;
    unsigned o0 = ((unsigned)(buf * 16 + cr0) * KPAD + cc4) * 4u;
    unsigned o1 = ((unsigned)(buf * 16 + cr1) * KPAD + cc4) * 4u;
    async_b128(ktbase + o0, ksrc + cr0 * HD_ + cc4);
    async_b128(ktbase + o1, ksrc + cr1 * HD_ + cc4);
    async_b128(vtbase + o0, vsrc + cr0 * HD_ + cc4);
    async_b128(vtbase + o1, vsrc + cr1 * HD_ + cc4);
  };

  // --- Q tile as 32 f32 A-fragments, pre-scaled ---
  v2f qa[32];
  {
    const float* qp =
        q + (size_t)(b * T_ + qbase + wave * 16 + lo) * (H_ * HD_) + h * HD_;
#pragma unroll
    for (int j = 0; j < 32; ++j) {
      v2f a = *reinterpret_cast<const v2f*>(qp + 4 * j + hi2);
      a.x *= QK_SCALE;
      a.y *= QK_SCALE;
      qa[j] = a;
    }
  }

  v8f oacc[8];
#pragma unroll
  for (int i = 0; i < 8; ++i) oacc[i] = v8zero();
  float mrun[8], lrun[8];
#pragma unroll
  for (int r = 0; r < 8; ++r) { mrun[r] = -1e30f; lrun[r] = 0.f; }

  const int nkb = qc * 8 + 8;
  issue_page(0, 0);
  for (int kb = 0; kb < nkb; ++kb) {
    const int cur = kb & 1;
    if (kb + 1 < nkb) {
      issue_page(cur ^ 1, kb + 1);
      asm volatile("s_wait_asynccnt 0x4" ::: "memory");  // drain current page
    } else {
      asm volatile("s_wait_asynccnt 0x0" ::: "memory");
    }
    __syncthreads();

    // --- S = Q @ K^T ---
    v8f s = v8zero();
#pragma unroll
    for (int j = 0; j < 32; ++j) {
      v2f bf = *reinterpret_cast<const v2f*>(&Kt[cur][lo][4 * j + hi2]);
      s = wmma_f32(qa[j], bf, s);
    }

    // --- causal mask + online softmax ---
    const int kcol = kb * 16 + lo;
    float rmax[8];
#pragma unroll
    for (int r = 0; r < 8; ++r) {
      int qpos = qbase + wave * 16 + r + hrow;
      float sv = s[r];
      if (qpos < kcol) sv = -1e9f;
      s[r] = sv;
      rmax[r] = sv;
    }
#pragma unroll
    for (int msk = 1; msk <= 8; msk <<= 1)
#pragma unroll
      for (int r = 0; r < 8; ++r)
        rmax[r] = fmaxf(rmax[r], __shfl_xor(rmax[r], msk, 32));

    float pscale[8];
#pragma unroll
    for (int r = 0; r < 8; ++r) {
      float mn = fmaxf(mrun[r], rmax[r]);
      pscale[r] = __expf(mrun[r] - mn);
      mrun[r] = mn;
    }
    float rsum[8];
#pragma unroll
    for (int r = 0; r < 8; ++r) {
      float p = __expf(s[r] - mrun[r]);
      s[r] = p;
      rsum[r] = p;
    }
#pragma unroll
    for (int msk = 1; msk <= 8; msk <<= 1)
#pragma unroll
      for (int r = 0; r < 8; ++r)
        rsum[r] += __shfl_xor(rsum[r], msk, 32);
#pragma unroll
    for (int r = 0; r < 8; ++r) lrun[r] = lrun[r] * pscale[r] + rsum[r];
#pragma unroll
    for (int ct = 0; ct < 8; ++ct)
#pragma unroll
      for (int r = 0; r < 8; ++r) oacc[ct][r] *= pscale[r];

    // --- P (C-layout) -> LDS -> A-fragments ---
#pragma unroll
    for (int r = 0; r < 8; ++r) Ps[wave][r + hrow][lo] = s[r];
    v2f pa[4];
#pragma unroll
    for (int j = 0; j < 4; ++j)
      pa[j] = *reinterpret_cast<const v2f*>(&Ps[wave][lo][4 * j + hi2]);

    // --- O += P @ V ---
#pragma unroll
    for (int ct = 0; ct < 8; ++ct) {
      v8f acc = oacc[ct];
#pragma unroll
      for (int j = 0; j < 4; ++j) {
        v2f bv;
        bv.x = Vt[cur][4 * j + hi2 + 0][ct * 16 + lo];
        bv.y = Vt[cur][4 * j + hi2 + 1][ct * 16 + lo];
        acc = wmma_f32(pa[j], bv, acc);
      }
      oacc[ct] = acc;
    }
    __syncthreads();  // all waves done with buf `cur` before it is re-filled
  }

  float invl[8];
#pragma unroll
  for (int r = 0; r < 8; ++r) invl[r] = 1.0f / lrun[r];
#pragma unroll
  for (int ct = 0; ct < 8; ++ct) {
    int col = h * HD_ + ct * 16 + lo;
#pragma unroll
    for (int r = 0; r < 8; ++r) {
      int row = b * T_ + qbase + wave * 16 + r + hrow;
      o[(size_t)row * (H_ * HD_) + col] = oacc[ct][r] * invl[r];
    }
  }
}

// ---------------------------------------------------------------------------
// Launch
// ---------------------------------------------------------------------------
extern "C" void kernel_launch(void* const* d_in, const int* in_sizes, int n_in,
                              void* d_out, int out_size, void* d_ws,
                              size_t ws_size, hipStream_t stream) {
  (void)in_sizes; (void)n_in; (void)out_size; (void)ws_size;
  const float* x        = (const float*)d_in[0];
  const float* w_q      = (const float*)d_in[1];
  const float* w_kv     = (const float*)d_in[2];
  const float* w_o      = (const float*)d_in[3];
  const float* rope_cos = (const float*)d_in[6];
  const float* rope_sin = (const float*)d_in[7];
  const int*   positions = (const int*)d_in[8];
  const int*   btab      = (const int*)d_in[9];

  char* ws = (char*)d_ws;
  float* q_buf    = (float*)(ws);                        // 64 MB
  float* kv_buf   = (float*)(ws + ((size_t)64 << 20));   // 32 MB
  float* attn_out = (float*)(ws + ((size_t)96 << 20));   // 64 MB
  float* k_ws     = (float*)(ws + ((size_t)160 << 20));  // 16 MB
  float* v_ws     = (float*)(ws + ((size_t)176 << 20));  // 16 MB

  dim3 blk(256);
  gemm_bf16x3<<<dim3(NTOK_ / GM, DM_ / GN), blk, 0, stream>>>(
      x, w_q, q_buf, NTOK_, DM_, DM_);
  gemm_bf16x3<<<dim3(NTOK_ / GM, (2 * KVH_ * HD_) / GN), blk, 0, stream>>>(
      x, w_kv, kv_buf, NTOK_, 2 * KVH_ * HD_, DM_);
  rope_scatter<<<dim3(NTOK_), blk, 0, stream>>>(
      q_buf, kv_buf, k_ws, v_ws, rope_cos, rope_sin, positions, btab);
  attn_fa<<<dim3(B_ * H_ * (T_ / 128)), blk, 0, stream>>>(
      q_buf, k_ws, v_ws, btab, attn_out);
  gemm_bf16x3<<<dim3(NTOK_ / GM, DM_ / GN), blk, 0, stream>>>(
      attn_out, w_o, (float*)d_out, NTOK_, DM_, DM_);
}
